// Window_Block_56169582297330
// MI455X (gfx1250) — matrix operand order repaired and checked by reading, not compiled, not run on hardware
//
#include <hip/hip_runtime.h>
#include <hip/hip_bf16.h>

typedef __bf16 bf16_t;
typedef unsigned int u32;
typedef bf16_t v16bf __attribute__((ext_vector_type(16)));
typedef float  v8f   __attribute__((ext_vector_type(8)));
typedef u32    v8u   __attribute__((ext_vector_type(8)));

__device__ __forceinline__ v8f wmma_bf16(v16bf a, v16bf b, v8f c) {
    return __builtin_amdgcn_wmma_f32_16x16x32_bf16(
        false, a, false, b, (short)0, c, false, false);
}

__device__ __forceinline__ v8f zero8() {
    v8f z;
#pragma unroll
    for (int i = 0; i < 8; ++i) z[i] = 0.0f;
    return z;
}

// ---- packed-pair fragment loaders (each VGPR = one aligned b32 load) ----
__device__ __forceinline__ v16bf load_a_pairs(const u32* row_w, int hi) {
    v8u u;
#pragma unroll
    for (int j = 0; j < 4; ++j) u[j] = row_w[hi * 4 + j];
#pragma unroll
    for (int j = 0; j < 4; ++j) u[4 + j] = row_w[8 + hi * 4 + j];
    return __builtin_bit_cast(v16bf, u);
}
__device__ __forceinline__ v16bf load_a_pairs_lo(const u32* row_w, int hi) {
    v8u u;
#pragma unroll
    for (int j = 0; j < 4; ++j) u[j] = row_w[hi * 4 + j];
#pragma unroll
    for (int j = 0; j < 4; ++j) u[4 + j] = 0u;
    return __builtin_bit_cast(v16bf, u);
}
__device__ __forceinline__ v16bf load_b_pairs(const u32* col_w, int hi, int stride) {
    v8u u;
#pragma unroll
    for (int j = 0; j < 8; ++j) u[j] = col_w[((j >> 2) * 8 + hi * 4 + (j & 3)) * stride];
    return __builtin_bit_cast(v16bf, u);
}
__device__ __forceinline__ v16bf load_b_contig(const u32* row_w, int hi) {
    v8u u;
#pragma unroll
    for (int j = 0; j < 8; ++j) u[j] = row_w[(j >> 2) * 8 + hi * 4 + (j & 3)];
    return __builtin_bit_cast(v16bf, u);
}
__device__ __forceinline__ v16bf load_b_contig_lo(const u32* row_w, int hi) {
    v8u u;
#pragma unroll
    for (int j = 0; j < 8; ++j) u[j] = (j < 4) ? row_w[hi * 4 + j] : 0u;
    return __builtin_bit_cast(v16bf, u);
}

__device__ __forceinline__ u32 pack2(float lo, float hi) {
    unsigned short a = __builtin_bit_cast(unsigned short, (bf16_t)lo);
    unsigned short b = __builtin_bit_cast(unsigned short, (bf16_t)hi);
    return (u32)a | ((u32)b << 16);
}

// Async global->LDS b128 copy (ASYNCcnt-tracked, bypasses VGPRs).
// lds_off: LDS byte address (flat addr truncated per ISA 10.2); gaddr: global.
__device__ __forceinline__ void async_copy_b128(u32 lds_off, const void* gaddr) {
    asm volatile("global_load_async_to_lds_b128 %0, %1, off"
                 :: "v"(lds_off), "v"(gaddr) : "memory");
}
__device__ __forceinline__ void wait_asynccnt0() {
    asm volatile("s_wait_asynccnt 0x0" ::: "memory");
}

#define LN_EPS 1e-5f
#define BN_EPS 1e-5f

// ---------------------------------------------------------------------------
// Weight prep: bf16 K-pair-packed u32 layouts for WMMA B-fragments.
// ---------------------------------------------------------------------------
__global__ void prep_weights(const float* __restrict__ qkv_w,
                             const float* __restrict__ w1,
                             const float* __restrict__ w2,
                             u32* __restrict__ wqkvp,
                             u32* __restrict__ wT1p,
                             u32* __restrict__ wT2p) {
    int i = blockIdx.x * 256 + threadIdx.x;
    if (i < 24576) {
        int kp = i / 384, n = i % 384;
        wqkvp[i] = pack2(qkv_w[n * 128 + 2 * kp], qkv_w[n * 128 + 2 * kp + 1]);
    } else if (i < 24576 + 73728) {
        int j = i - 24576;
        int tap = j >> 13, rem = j & 8191, kp = rem >> 7, o = rem & 127;
        wT1p[j] = pack2(w1[(o * 128 + 2 * kp) * 9 + tap],
                        w1[(o * 128 + 2 * kp + 1) * 9 + tap]);
    } else if (i < 24576 + 2 * 73728) {
        int j = i - 24576 - 73728;
        int tap = j >> 13, rem = j & 8191, kp = rem >> 7, o = rem & 127;
        wT2p[j] = pack2(w2[(o * 128 + 2 * kp) * 9 + tap],
                        w2[(o * 128 + 2 * kp + 1) * 9 + tap]);
    }
}

// ---------------------------------------------------------------------------
// Window attention: 1 WG (8 waves) per 8x8 window; wave w = head w.
// ---------------------------------------------------------------------------
__global__ __launch_bounds__(256) void attn_kernel(
    const float* __restrict__ x,
    const float* __restrict__ n1w, const float* __restrict__ n1b,
    const u32* __restrict__ wqkvp,               // [64 kp][384]
    const float* __restrict__ bias_table,        // [225][8]
    const float* __restrict__ n2w, const float* __restrict__ n2b,
    float* __restrict__ attn_out) {
    __shared__ __align__(16) unsigned char pool[118016];
    float*  winf = (float*)(pool);                 // stride 129
    bf16_t* vtb  = (bf16_t*)(pool + 33024);        // vT[d*66 + tok]
    bf16_t* hbf  = (bf16_t*)(pool + 49920);        // stride 130
    bf16_t* qbf  = (bf16_t*)(pool + 66560);        // stride 130
    bf16_t* kbf  = (bf16_t*)(pool + 83200);        // stride 130
    float*  mu_s = (float*)(pool + 117504);
    float*  rs_s = (float*)(pool + 117504 + 256);

    const int tid = threadIdx.x;
    const int lane = tid & 31, wv = tid >> 5;
    const int ln = lane & 15, hi = lane >> 4;
    const int bi = blockIdx.x >> 8, wid = blockIdx.x & 255;
    const int wh = wid >> 4, ww = wid & 15;

    for (int e = tid; e < 8192; e += 256) {
        int c = e >> 6, t = e & 63;
        int gy = wh * 8 + (t >> 3), gx = ww * 8 + (t & 7);
        winf[t * 129 + c] = x[(((size_t)bi * 128 + c) * 128 + gy) * 128 + gx];
    }
    __syncthreads();

    if (tid < 64) {
        float sm = 0.f, sq = 0.f;
        for (int c = 0; c < 128; ++c) {
            float t = winf[tid * 129 + c];
            sm += t; sq += t * t;
        }
        float mu = sm * (1.0f / 128.0f);
        float var = sq * (1.0f / 128.0f) - mu * mu;
        mu_s[tid] = mu;
        rs_s[tid] = rsqrtf(var + LN_EPS);
    }
    __syncthreads();
    for (int e = tid; e < 8192; e += 256) {
        int t = e >> 7, c = e & 127;
        float h = (winf[t * 129 + c] - mu_s[t]) * rs_s[t] * n1w[c] + n1b[c];
        hbf[t * 130 + c] = (bf16_t)h;
    }
    __syncthreads();

    // ---- QKV GEMM: [64x128] x [128x384]; wave owns 3 N-tiles ----
    v8f qacc[3][4];
#pragma unroll
    for (int j = 0; j < 3; ++j)
#pragma unroll
        for (int mt = 0; mt < 4; ++mt) qacc[j][mt] = zero8();

    for (int ks = 0; ks < 4; ++ks) {
        v16bf afr[4];
#pragma unroll
        for (int mt = 0; mt < 4; ++mt)
            afr[mt] = load_a_pairs(
                (const u32*)(hbf + (mt * 16 + ln) * 130 + ks * 32), hi);
#pragma unroll
        for (int j = 0; j < 3; ++j) {
            int n = (wv * 3 + j) * 16 + ln;
            v16bf bfr = load_b_pairs(wqkvp + (ks * 16) * 384 + n, hi, 384);
#pragma unroll
            for (int mt = 0; mt < 4; ++mt)
                qacc[j][mt] = wmma_bf16(afr[mt], bfr, qacc[j][mt]);
        }
    }
#pragma unroll
    for (int j = 0; j < 3; ++j) {
        int ncol = (wv * 3 + j) * 16 + ln;
        int which = ncol >> 7, d = ncol & 127;
#pragma unroll
        for (int mt = 0; mt < 4; ++mt)
#pragma unroll
            for (int i = 0; i < 8; ++i) {
                int m = mt * 16 + i + hi * 8;
                float v = qacc[j][mt][i];
                if (which == 0)      qbf[m * 130 + d] = (bf16_t)v;
                else if (which == 1) kbf[m * 130 + d] = (bf16_t)v;
                else                 vtb[d * 66 + m] = (bf16_t)v;
            }
    }
    __syncthreads();

    // ---- L2 normalize q,k per (token, head) ----
    for (int task = tid; task < 1024; task += 256) {
        int which = task >> 9, t = (task >> 3) & 63, hh = task & 7;
        bf16_t* base = (which ? kbf : qbf) + t * 130 + hh * 16;
        float ss = 0.f;
#pragma unroll
        for (int d = 0; d < 16; ++d) { float q = (float)base[d]; ss += q * q; }
        float inv = rsqrtf(ss);
#pragma unroll
        for (int d = 0; d < 16; ++d) base[d] = (bf16_t)((float)base[d] * inv);
    }
    __syncthreads();

    // ---- scores: per-head 64x64, K=16 zero-padded to 32 ----
    const int h = wv;
    v8f s[4][4];
#pragma unroll
    for (int mt = 0; mt < 4; ++mt)
#pragma unroll
        for (int nt = 0; nt < 4; ++nt) s[mt][nt] = zero8();
    {
        v16bf afr[4];
#pragma unroll
        for (int mt = 0; mt < 4; ++mt)
            afr[mt] = load_a_pairs_lo(
                (const u32*)(qbf + (mt * 16 + ln) * 130 + h * 16), hi);
#pragma unroll
        for (int nt = 0; nt < 4; ++nt) {
            v16bf bfr = load_b_contig_lo(
                (const u32*)(kbf + (nt * 16 + ln) * 130 + h * 16), hi);
#pragma unroll
            for (int mt = 0; mt < 4; ++mt)
                s[mt][nt] = wmma_bf16(afr[mt], bfr, s[mt][nt]);
        }
    }
    __syncthreads();   // q,k dead -> probs may overlay

    bf16_t* pb = (bf16_t*)(pool + 49920) + h * 4224;   // [64][66] per head

#pragma unroll
    for (int mt = 0; mt < 4; ++mt) {
#pragma unroll
        for (int i = 0; i < 8; ++i) {
            int m = mt * 16 + i + hi * 8;
            int ri = m >> 3, ci = m & 7;
            float vals[4];
#pragma unroll
            for (int nt = 0; nt < 4; ++nt) {
                int n = nt * 16 + ln;
                int rj = n >> 3, cj = n & 7;
                int idx = (ri - rj + 7) * 15 + (ci - cj + 7);
                vals[nt] = s[mt][nt][i] * 0.25f + bias_table[idx * 8 + h];
            }
            float mx = fmaxf(fmaxf(vals[0], vals[1]), fmaxf(vals[2], vals[3]));
#pragma unroll
            for (int off = 1; off < 16; off <<= 1)
                mx = fmaxf(mx, __shfl_xor(mx, off, 32));
            float sum = 0.f;
#pragma unroll
            for (int nt = 0; nt < 4; ++nt) {
                vals[nt] = __expf(vals[nt] - mx);
                sum += vals[nt];
            }
#pragma unroll
            for (int off = 1; off < 16; off <<= 1)
                sum += __shfl_xor(sum, off, 32);
            float inv = 1.0f / sum;
#pragma unroll
            for (int nt = 0; nt < 4; ++nt)
                pb[m * 66 + nt * 16 + ln] = (bf16_t)(vals[nt] * inv);
        }
    }

    // ---- O = probs[64x64] x V[64x16] ----
    v8f oacc[4];
#pragma unroll
    for (int mt = 0; mt < 4; ++mt) oacc[mt] = zero8();
#pragma unroll
    for (int ks = 0; ks < 2; ++ks) {
        v16bf bfr = load_b_contig(
            (const u32*)(vtb + (h * 16 + ln) * 66 + ks * 32), hi);
#pragma unroll
        for (int mt = 0; mt < 4; ++mt) {
            v16bf afr = load_a_pairs(
                (const u32*)(pb + (mt * 16 + ln) * 66 + ks * 32), hi);
            oacc[mt] = wmma_bf16(afr, bfr, oacc[mt]);
        }
    }

#pragma unroll
    for (int mt = 0; mt < 4; ++mt)
#pragma unroll
        for (int i = 0; i < 8; ++i) {
            int m = mt * 16 + i + hi * 8;
            int c0 = h * 16 + ln;
            winf[m * 129 + c0] = winf[m * 129 + c0] + oacc[mt][i];
        }
    __syncthreads();

    if (tid < 64) {
        float sm = 0.f, sq = 0.f;
        for (int c = 0; c < 128; ++c) {
            float t = winf[tid * 129 + c];
            sm += t; sq += t * t;
        }
        float mu = sm * (1.0f / 128.0f);
        float var = sq * (1.0f / 128.0f) - mu * mu;
        mu_s[tid] = mu;
        rs_s[tid] = rsqrtf(var + LN_EPS);
    }
    __syncthreads();
    for (int e = tid; e < 8192; e += 256) {
        int c = e >> 6, t = e & 63;
        int gy = wh * 8 + (t >> 3), gx = ww * 8 + (t & 7);
        float val = (winf[t * 129 + c] - mu_s[t]) * rs_s[t] * n2w[c] + n2b[c];
        attn_out[(((size_t)bi * 128 + c) * 128 + gy) * 128 + gx] = val;
    }
}

// ---------------------------------------------------------------------------
// Conv3x3(128->128) + BN + exact GELU [+ residual], implicit GEMM.
// Weight tiles streamed global->LDS with GLOBAL_LOAD_ASYNC_TO_LDS_B128.
// ---------------------------------------------------------------------------
template <bool IN_BF16, bool RESID>
__global__ __launch_bounds__(256) void conv_bn_gelu_kernel(
    const void* __restrict__ in_, const u32* __restrict__ wTp,
    const float* __restrict__ bng, const float* __restrict__ bnb,
    const float* __restrict__ bnm, const float* __restrict__ bnv,
    const float* __restrict__ resid, void* __restrict__ out_) {
    __shared__ __align__(16) unsigned char pool[117008];
    bf16_t* patch  = (bf16_t*)(pool);           // [(py*18+px)*130 + c]
    u32*    wtap_w = (u32*)(pool + 84240);      // [kp*128 + o]
    float*  stage  = (float*)(pool);            // [pl*129 + oc] (epilogue)

    const int tid = threadIdx.x;
    const int lane = tid & 31, wv = tid >> 5;
    const int ln = lane & 15, hi = lane >> 4;
    const int bx = blockIdx.x;
    const int bi = bx >> 6, t = bx & 63;
    const int ty0 = (t >> 3) * 16, tx0 = (t & 7) * 16;
    const float* inF = (const float*)in_;
    const bf16_t* inB = (const bf16_t*)in_;
    const u32 wtap_lds = (u32)(uintptr_t)wtap_w;   // LDS byte address

    // ---- coalesced halo-tile load: wave reads an 18-wide row per iter ----
    for (int r = wv; r < 18 * 128; r += 8) {
        int py = r % 18, c = r / 18;
        int gy = ty0 + py - 1;
        if (lane < 18) {
            int gx = tx0 + lane - 1;
            float v = 0.f;
            if (gy >= 0 && gy < 128 && gx >= 0 && gx < 128) {
                size_t gi = (((size_t)bi * 128 + c) * 128 + gy) * 128 + gx;
                v = IN_BF16 ? (float)inB[gi] : inF[gi];
            }
            patch[(py * 18 + lane) * 130 + c] = (bf16_t)v;
        }
    }

    v8f acc[2][8];
#pragma unroll
    for (int a = 0; a < 2; ++a)
#pragma unroll
        for (int n = 0; n < 8; ++n) acc[a][n] = zero8();

#pragma unroll 1
    for (int tap = 0; tap < 9; ++tap) {
        __syncthreads();
        // async 32 KB weight-tile copy: 8 x b128 per thread
#pragma unroll
        for (int e = 0; e < 8; ++e) {
            int idx = tid + e * 256;                       // uint4 index
            async_copy_b128(wtap_lds + idx * 16,
                            wTp + tap * 8192 + idx * 4);
        }
        wait_asynccnt0();
        __syncthreads();
        if (tap < 8)   // prefetch next tap tile (global_prefetch_b8)
            __builtin_prefetch(wTp + (tap + 1) * 8192 + (tid & 63) * 128, 0, 1);
        int dy = tap / 3, dx = tap % 3;
#pragma unroll
        for (int ks = 0; ks < 4; ++ks) {
            v16bf afr[2];
#pragma unroll
            for (int m2 = 0; m2 < 2; ++m2) {
                int m = (wv * 2 + m2) * 16 + ln;
                int pr = m >> 4, pc = m & 15;
                afr[m2] = load_a_pairs(
                    (const u32*)(patch + ((pr + dy) * 18 + (pc + dx)) * 130 +
                                 ks * 32), hi);
            }
#pragma unroll
            for (int nt = 0; nt < 8; ++nt) {
                v16bf bfr =
                    load_b_pairs(wtap_w + (ks * 16) * 128 + nt * 16 + ln, hi, 128);
                acc[0][nt] = wmma_bf16(afr[0], bfr, acc[0][nt]);
                acc[1][nt] = wmma_bf16(afr[1], bfr, acc[1][nt]);
            }
        }
    }

    float pm[8], pi[8], pg[8], pb2[8];
#pragma unroll
    for (int nt = 0; nt < 8; ++nt) {
        int oc = nt * 16 + ln;
        pm[nt] = bnm[oc];
        pi[nt] = rsqrtf(bnv[oc] + BN_EPS);
        pg[nt] = bng[oc];
        pb2[nt] = bnb[oc];
    }
    __syncthreads();   // patch/wtap reads done; stage may overlay

#pragma unroll
    for (int half = 0; half < 2; ++half) {
        if ((wv >> 2) == half) {
#pragma unroll
            for (int m2 = 0; m2 < 2; ++m2)
#pragma unroll
                for (int nt = 0; nt < 8; ++nt)
#pragma unroll
                    for (int i = 0; i < 8; ++i) {
                        int m = (wv * 2 + m2) * 16 + i + hi * 8;
                        int pl = m & 127;
                        float val = acc[m2][nt][i];
                        val = (val - pm[nt]) * pi[nt] * pg[nt] + pb2[nt];
                        val = 0.5f * val * (1.0f + erff(val * 0.70710678118f));
                        stage[pl * 129 + nt * 16 + ln] = val;
                    }
        }
        __syncthreads();
        for (int e = tid; e < 16384; e += 256) {
            int pl = e & 127, oc = e >> 7;
            int m = half * 128 + pl;
            int gy = ty0 + (m >> 4), gx = tx0 + (m & 15);
            size_t gi = (((size_t)bi * 128 + oc) * 128 + gy) * 128 + gx;
            float val = stage[pl * 129 + oc];
            if (RESID)
                ((float*)out_)[gi] = val + resid[gi];
            else
                ((bf16_t*)out_)[gi] = (bf16_t)val;
        }
        __syncthreads();
    }
}

// ---------------------------------------------------------------------------
extern "C" void kernel_launch(void* const* d_in, const int* in_sizes, int n_in,
                              void* d_out, int out_size, void* d_ws, size_t ws_size,
                              hipStream_t stream) {
    const float* x          = (const float*)d_in[0];
    const float* norm1_w    = (const float*)d_in[1];
    const float* norm1_b    = (const float*)d_in[2];
    const float* qkv_w      = (const float*)d_in[3];
    const float* bias_table = (const float*)d_in[4];
    const float* norm2_w    = (const float*)d_in[5];
    const float* norm2_b    = (const float*)d_in[6];
    const float* conv1_w    = (const float*)d_in[7];
    const float* bn1_g      = (const float*)d_in[8];
    const float* bn1_b      = (const float*)d_in[9];
    const float* bn1_m      = (const float*)d_in[10];
    const float* bn1_v      = (const float*)d_in[11];
    const float* conv2_w    = (const float*)d_in[12];
    const float* bn2_g      = (const float*)d_in[13];
    const float* bn2_b      = (const float*)d_in[14];
    const float* bn2_m      = (const float*)d_in[15];
    const float* bn2_v      = (const float*)d_in[16];

    char* ws = (char*)d_ws;
    float*  attn_out = (float*)ws;                               // 67,108,864 B
    bf16_t* y1       = (bf16_t*)(ws + 67108864);                 // 33,554,432 B
    u32*    wqkvp    = (u32*)(ws + 67108864 + 33554432);         //     98,304 B
    u32*    wT1p     = (u32*)((char*)wqkvp + 98304);             //    294,912 B
    u32*    wT2p     = (u32*)((char*)wT1p + 294912);             //    294,912 B

    prep_weights<<<672, 256, 0, stream>>>(qkv_w, conv1_w, conv2_w,
                                          wqkvp, wT1p, wT2p);

    attn_kernel<<<2048, 256, 0, stream>>>(x, norm1_w, norm1_b, wqkvp,
                                          bias_table, norm2_w, norm2_b,
                                          attn_out);

    conv_bn_gelu_kernel<false, false><<<512, 256, 0, stream>>>(
        attn_out, wT1p, bn1_g, bn1_b, bn1_m, bn1_v, nullptr, y1);

    conv_bn_gelu_kernel<true, true><<<512, 256, 0, stream>>>(
        y1, wT2p, bn2_g, bn2_b, bn2_m, bn2_v, attn_out, (float*)d_out);
}
